// propagation_backward_18305150615731
// MI455X (gfx1250) — matrix operand order repaired and checked by reading, not compiled, not run on hardware
//
#include <hip/hip_runtime.h>
#include <hip/hip_bf16.h>

#define HH 192
#define WW 192
#define NF 64
#define HW (HH * WW)
#define PW (WW + 2)                 // padded row stride (pixels)
#define NP ((HH + 2) * (WW + 2))    // padded pixel count (halo of 1)

typedef __bf16 bf16_t;
typedef __attribute__((ext_vector_type(16))) __bf16 v16bf;
typedef __attribute__((ext_vector_type(8)))  float  v8f;
typedef __attribute__((ext_vector_type(8)))  int    v8i;
typedef __attribute__((ext_vector_type(4)))  int    v4i;

union ARegs  { v4i half4[2]; v16bf b; };   // 32 bytes
union BRegs  { v8i i;        v16bf b; };   // 32 bytes
union OutPack{ v4i i;        __bf16 h[8]; };

// ---------------------------------------------------------------------------
// NCHW f32 -> padded NHWC bf16 (interior only; halo kept zero by memset)
// ---------------------------------------------------------------------------
extern "C" __global__ void cvt_nchw_f32_to_nhwc_bf16(const float* __restrict__ src,
                                                     bf16_t* __restrict__ dst, int C)
{
    int idx = blockIdx.x * blockDim.x + threadIdx.x;
    int total = C * HW;
    if (idx >= total) return;
    int c = idx / HW;
    int p = idx - c * HW;
    int y = p / WW, x = p - y * WW;
    dst[((size_t)(y + 1) * PW + (x + 1)) * C + c] = (__bf16)src[idx];
}

// ---------------------------------------------------------------------------
// Weight pack: (Cout, Cin, 3, 3) f32 -> [Cout][ tap*Cin + ci ] bf16 (tap-major)
// ---------------------------------------------------------------------------
extern "C" __global__ void pack_weights_bf16(const float* __restrict__ w,
                                             bf16_t* __restrict__ dst,
                                             int Cout, int Cin)
{
    int idx = blockIdx.x * blockDim.x + threadIdx.x;
    int total = Cout * Cin * 9;
    if (idx >= total) return;
    int co  = idx / (Cin * 9);
    int r   = idx - co * (Cin * 9);
    int ci  = r / 9;
    int tap = r - ci * 9;
    dst[(size_t)co * (9 * Cin) + tap * Cin + ci] = (__bf16)w[idx];
}

// ---------------------------------------------------------------------------
// Implicit-GEMM 3x3 SAME conv on halo-padded NHWC bf16 tensors.
// 128 threads = 4 waves arranged 2(M) x 2(N): WG tile = 64 outch x 64 pixels.
// Each wave owns a 2x2 register tile (32 outch x 32 px, 4 accumulators), so
// every K-block is 8 x global_load_b128 feeding 4 x v_wmma (2 loads/WMMA).
// Fully unrolled: straight-line {A/B-loads, 4x v_wmma} sequence.
//   CTA/CTB: total channels of srcA/srcB, COFA: channel offset into srcA,
//   CINA/CINB: channels consumed (CINB=0 -> no concat), ACT: 0 none 1 lrelu,
//   CTO: out channel stride, COUT: total out channels (grid.y = COUT/64).
// ---------------------------------------------------------------------------
template<int CTA, int COFA, int CINA, int CTB, int CINB, int ACT, int CTO, int COUT>
__global__ __launch_bounds__(128)
void conv3x3_wmma(const bf16_t* __restrict__ srcA,
                  const bf16_t* __restrict__ srcB_,
                  const bf16_t* __restrict__ wts,
                  const float*  __restrict__ bias,
                  bf16_t* __restrict__ out)
{
    constexpr int CINT = CINA + CINB;
    constexpr int KD   = 9 * CINT;

    const int lane = threadIdx.x & 31;
    const int wave = threadIdx.x >> 5;
    const int mw   = wave & 1;                    // M sub-block within WG
    const int nw   = wave >> 1;                   // N sub-block within WG
    const int tile = blockIdx.x;                  // 192 rows * 3 tiles/row
    const int y    = tile / (WW / 64);
    const int x0   = (tile - y * (WW / 64)) * 64 + nw * 32;
    const int co0  = blockIdx.y * 64 + mw * 32;
    const int n    = lane & 15;                   // pixel / A-row index
    const int half = lane >> 4;                   // K-half selector
    const int px   = x0 + n;

    // compile-time fold: srcB only used when CINB > 0
    const bf16_t* srcB = (CINB > 0) ? srcB_ : srcA;

    v8f acc00 = {0.f,0.f,0.f,0.f,0.f,0.f,0.f,0.f};
    v8f acc01 = {0.f,0.f,0.f,0.f,0.f,0.f,0.f,0.f};
    v8f acc10 = {0.f,0.f,0.f,0.f,0.f,0.f,0.f,0.f};
    v8f acc11 = {0.f,0.f,0.f,0.f,0.f,0.f,0.f,0.f};

    // A rows: lanes 0-15 K {0..15}, lanes 16-31 K {16..31}
    const bf16_t* wrow = wts + (size_t)(co0 + n) * KD + half * 16;
    // B pixels: tap (ky,kx) of interior pixel (y,px) = padded pixel (y+ky, px+kx)
    const bf16_t* qA = srcA + ((size_t)y * PW + px) * CTA + COFA + half * 16;
    const bf16_t* qB = srcB + ((size_t)y * PW + px) * CTB + half * 16;

    auto mma_block = [&](const bf16_t* wk, const bf16_t* b0p, const bf16_t* b1p) {
        ARegs a0, a1; BRegs b0, b1;
        a0.half4[0] = *(const v4i*)(wk);
        a0.half4[1] = *(const v4i*)(wk + 8);
        a1.half4[0] = *(const v4i*)(wk + (size_t)16 * KD);
        a1.half4[1] = *(const v4i*)(wk + (size_t)16 * KD + 8);
        b0.i = *(const v8i*)(b0p);
        b1.i = *(const v8i*)(b1p);
        acc00 = __builtin_amdgcn_wmma_f32_16x16x32_bf16(false, a0.b, false, b0.b,
                                                        (short)0, acc00, false, false);
        acc01 = __builtin_amdgcn_wmma_f32_16x16x32_bf16(false, a0.b, false, b1.b,
                                                        (short)0, acc01, false, false);
        acc10 = __builtin_amdgcn_wmma_f32_16x16x32_bf16(false, a1.b, false, b0.b,
                                                        (short)0, acc10, false, false);
        acc11 = __builtin_amdgcn_wmma_f32_16x16x32_bf16(false, a1.b, false, b1.b,
                                                        (short)0, acc11, false, false);
    };

#pragma unroll
    for (int ky = 0; ky < 3; ++ky) {
#pragma unroll
        for (int kx = 0; kx < 3; ++kx) {
            const int tap = ky * 3 + kx;
            const bf16_t* wk = wrow + tap * CINT;
            const bf16_t* a0 = qA + ((size_t)ky * PW + kx) * CTA;
#pragma unroll
            for (int cb = 0; cb < CINA; cb += 32)
                mma_block(wk + cb, a0 + cb, a0 + (size_t)16 * CTA + cb);
            if constexpr (CINB > 0) {
                const bf16_t* b0 = qB + ((size_t)ky * PW + kx) * CTB;
#pragma unroll
                for (int cb = 0; cb < CINB; cb += 32)
                    mma_block(wk + CINA + cb, b0 + cb, b0 + (size_t)16 * CTB + cb);
            }
        }
    }

    // C/D layout: VGPR v -> M = v + half*8, N = lane&15
    auto epilogue = [&](const v8f& a, int mi, int ni) {
        const int cbase = co0 + mi * 16 + half * 8;
        OutPack op;
#pragma unroll
        for (int v = 0; v < 8; ++v) {
            float r = a[v] + bias[cbase + v];
            if (ACT == 1) r = (r >= 0.f) ? r : 0.1f * r;
            op.h[v] = (__bf16)r;
        }
        *(v4i*)(out + ((size_t)(y + 1) * PW + (px + ni * 16 + 1)) * CTO + cbase) = op.i;
    };
    epilogue(acc00, 0, 0);
    epilogue(acc01, 0, 1);
    epilogue(acc10, 1, 0);
    epilogue(acc11, 1, 1);
}

// ---------------------------------------------------------------------------
// Per-pixel dynamic 3x3 kernel conv (all tensors halo-padded NHWC):
//   out[p,c] = sum_tap kp[p, tap*64+c] * feat_prop[p + tap_offset, c]
// ---------------------------------------------------------------------------
extern "C" __global__ void dyn_kernel_conv(const bf16_t* __restrict__ fp,
                                           const bf16_t* __restrict__ kp,
                                           bf16_t* __restrict__ outp)
{
    int idx = blockIdx.x * blockDim.x + threadIdx.x;
    if (idx >= HW * NF) return;
    int c = idx & (NF - 1);
    int p = idx >> 6;
    int y = p / WW, x = p - y * WW;
    const size_t pc = (size_t)(y + 1) * PW + (x + 1);
    float s = 0.f;
#pragma unroll
    for (int tap = 0; tap < 9; ++tap) {
        const int ky = tap / 3, kx = tap % 3;
        float fv = (float)fp[((size_t)(y + ky) * PW + (x + kx)) * NF + c];
        s += (float)kp[pc * (9 * NF) + tap * NF + c] * fv;
    }
    outp[pc * NF + c] = (__bf16)s;
}

// ---------------------------------------------------------------------------
// Gating combine: r = p1*sigmoid(g1) + p2*sigmoid(g2)
// Writes next feat_prop (padded NHWC bf16) and the f32 NCHW output slice.
// ---------------------------------------------------------------------------
extern "C" __global__ void gate_combine(const bf16_t* __restrict__ fusion,
                                        const bf16_t* __restrict__ g1,
                                        const bf16_t* __restrict__ g2,
                                        bf16_t* __restrict__ fp_out,
                                        float* __restrict__ dout)
{
    int idx = blockIdx.x * blockDim.x + threadIdx.x;
    if (idx >= HW * NF) return;
    int c = idx & (NF - 1);
    int p = idx >> 6;
    int y = p / WW, x = p - y * WW;
    const size_t pc = (size_t)(y + 1) * PW + (x + 1);
    float p1 = (float)fusion[pc * 128 + c];
    float p2 = (float)fusion[pc * 128 + 64 + c];
    float a  = (float)g1[pc * NF + c];
    float b  = (float)g2[pc * NF + c];
    float r  = p1 * (1.f / (1.f + __expf(-a))) + p2 * (1.f / (1.f + __expf(-b)));
    fp_out[pc * NF + c] = (__bf16)r;
    dout[(size_t)c * HW + p] = r;               // NCHW f32
}

// ---------------------------------------------------------------------------
// Host orchestration
// ---------------------------------------------------------------------------
extern "C" void kernel_launch(void* const* d_in, const int* in_sizes, int n_in,
                              void* d_out, int out_size, void* d_ws, size_t ws_size,
                              hipStream_t stream)
{
    const float* feature = (const float*)d_in[0];
    const float* w_kc1   = (const float*)d_in[1];
    const float* b_kc1   = (const float*)d_in[2];
    const float* w_kc2   = (const float*)d_in[3];
    const float* b_kc2   = (const float*)d_in[4];
    const float* w_c1    = (const float*)d_in[5];
    const float* b_c1    = (const float*)d_in[6];
    const float* w_c2    = (const float*)d_in[7];
    const float* b_c2    = (const float*)d_in[8];
    const float* w_c3    = (const float*)d_in[9];
    const float* b_c3    = (const float*)d_in[10];
    float* out = (float*)d_out;

    char* ws = (char*)d_ws;
    size_t off = 0;
    auto alloc = [&](size_t bytes) -> void* {
        void* p = ws + off;
        off += (bytes + 255) & ~(size_t)255;
        return p;
    };

    bf16_t* xfeat  = (bf16_t*)alloc((size_t)NP * 64  * 2);
    bf16_t* fp_cur = (bf16_t*)alloc((size_t)NP * 64  * 2);
    bf16_t* fp_new = (bf16_t*)alloc((size_t)NP * 64  * 2);
    bf16_t* tmp1   = (bf16_t*)alloc((size_t)NP * 64  * 2);
    bf16_t* fusion = (bf16_t*)alloc((size_t)NP * 128 * 2);
    bf16_t* g1     = (bf16_t*)alloc((size_t)NP * 64  * 2);
    bf16_t* g2     = (bf16_t*)alloc((size_t)NP * 64  * 2);
    bf16_t* kp     = (bf16_t*)alloc((size_t)NP * 576 * 2);
    bf16_t* w_kc1p = (bf16_t*)alloc((size_t)64  * 1152 * 2);
    bf16_t* w_kc2p = (bf16_t*)alloc((size_t)576 * 576  * 2);
    bf16_t* w_c1p  = (bf16_t*)alloc((size_t)128 * 1152 * 2);
    bf16_t* w_c2p  = (bf16_t*)alloc((size_t)64  * 576  * 2);
    bf16_t* w_c3p  = (bf16_t*)alloc((size_t)64  * 576  * 2);

    const int EB = 256;
    auto eblk = [&](int n) { return (n + EB - 1) / EB; };

    // zero halos (interiors are rewritten every step -> deterministic)
    hipMemsetAsync(xfeat,  0, (size_t)NP * 64  * 2, stream);
    hipMemsetAsync(fp_cur, 0, (size_t)NP * 64  * 2, stream);
    hipMemsetAsync(fp_new, 0, (size_t)NP * 64  * 2, stream);
    hipMemsetAsync(tmp1,   0, (size_t)NP * 64  * 2, stream);
    hipMemsetAsync(fusion, 0, (size_t)NP * 128 * 2, stream);

    // pack weights (deterministic, redone each call)
    pack_weights_bf16<<<eblk(64  * 128 * 9), EB, 0, stream>>>(w_kc1, w_kc1p, 64, 128);
    pack_weights_bf16<<<eblk(576 * 64  * 9), EB, 0, stream>>>(w_kc2, w_kc2p, 576, 64);
    pack_weights_bf16<<<eblk(128 * 128 * 9), EB, 0, stream>>>(w_c1,  w_c1p, 128, 128);
    pack_weights_bf16<<<eblk(64  * 64  * 9), EB, 0, stream>>>(w_c2,  w_c2p, 64, 64);
    pack_weights_bf16<<<eblk(64  * 64  * 9), EB, 0, stream>>>(w_c3,  w_c3p, 64, 64);

    // last frame: passthrough output + initial feat_prop
    hipMemcpyAsync(out + (size_t)9 * 64 * HW, feature + (size_t)9 * 64 * HW,
                   (size_t)64 * HW * sizeof(float), hipMemcpyDeviceToDevice, stream);
    cvt_nchw_f32_to_nhwc_bf16<<<eblk(64 * HW), EB, 0, stream>>>(
        feature + (size_t)9 * 64 * HW, fp_cur, 64);

    const dim3 cblock(128);
    const int  ptiles = HH * (WW / 64);   // 576 pixel tiles (64 px each)

    for (int i = 8; i >= 0; --i) {
        // x_feat -> padded NHWC bf16
        cvt_nchw_f32_to_nhwc_bf16<<<eblk(64 * HW), EB, 0, stream>>>(
            feature + (size_t)i * 64 * HW, xfeat, 64);

        // kc1: concat(x_feat, feat_prop) 128 -> 64, lrelu
        conv3x3_wmma<64, 0, 64, 64, 64, 1, 64, 64>
            <<<dim3(ptiles, 1), cblock, 0, stream>>>(xfeat, fp_cur, w_kc1p, b_kc1, tmp1);

        // kc2: 64 -> 576 (kernel_predict)
        conv3x3_wmma<64, 0, 64, 0, 0, 0, 576, 576>
            <<<dim3(ptiles, 9), cblock, 0, stream>>>(tmp1, nullptr, w_kc2p, b_kc2, kp);

        // dynamic per-pixel kernel conv
        dyn_kernel_conv<<<eblk(HW * NF), EB, 0, stream>>>(fp_cur, kp, fp_new);

        // c1: concat(x_feat, feat_prop) 128 -> 128, lrelu
        conv3x3_wmma<64, 0, 64, 64, 64, 1, 128, 128>
            <<<dim3(ptiles, 2), cblock, 0, stream>>>(xfeat, fp_new, w_c1p, b_c1, fusion);

        // c2 / c3: 64 -> 64 gating convs on the two halves of fusion
        conv3x3_wmma<128, 0, 64, 0, 0, 0, 64, 64>
            <<<dim3(ptiles, 1), cblock, 0, stream>>>(fusion, nullptr, w_c2p, b_c2, g1);
        conv3x3_wmma<128, 64, 64, 0, 0, 0, 64, 64>
            <<<dim3(ptiles, 1), cblock, 0, stream>>>(fusion, nullptr, w_c3p, b_c3, g2);

        // gating + write output slice i and next feat_prop
        gate_combine<<<eblk(HW * NF), EB, 0, stream>>>(
            fusion, g1, g2, fp_cur, out + (size_t)i * 64 * HW);
    }
}
